// EncoderLayer_23132693856802
// MI455X (gfx1250) — matrix operand order repaired and checked by reading, not compiled, not run on hardware
//
#include <hip/hip_runtime.h>

// ---------------------------------------------------------------------------
// Types for CDNA5 WMMA (wave32)
// ---------------------------------------------------------------------------
typedef __bf16 bf16;
typedef __attribute__((ext_vector_type(16))) __bf16 v16bf;
typedef __attribute__((ext_vector_type(8)))  __bf16 v8bf;
typedef __attribute__((ext_vector_type(8)))  float  v8f;

struct bf16x16 { v8bf lo; v8bf hi; };

#define B_   8
#define S_   1024
#define D_   512
#define H_   8
#define DK_  64
#define DFF_ 2048
#define BS_  (B_ * S_)       // 8192
#define SCALE_ 0.125f        // 1/sqrt(64)
#define EPS_   1e-5f

// GEMM epilogue flags
#define GF_OUTBF 1
#define GF_RELU  2
#define GF_TRANS 4

// ---------------------------------------------------------------------------
// Generic bf16 WMMA GEMM:  C[M,N] (+epilogue) = A(MxK bf16 rm) * Bt(NxK bf16 rm)^T
// Batched over grid.z; per-z offsets = b*(_B) + h*(_H), z = b*Hh + h.
// Each wave: 32 rows x 64 cols -> 8 x v_wmma_f32_16x16x32_bf16 per K-step
// from 2 A-fragments + 4 B-fragments (12 b128 loads -> 8 WMMAs).
// Block = 128 threads = 4 waves covering a 128x64 tile.
// Requires: M%128==0, N%64==0, K%32==0 (true for every GEMM in this model).
// ---------------------------------------------------------------------------
__global__ __launch_bounds__(128)
void gemm_bf16_wmma(const bf16* __restrict__ A, int lda, long long aB, long long aH,
                    const bf16* __restrict__ Bt, int ldb, long long bB, long long bH,
                    void* __restrict__ Cv, int ldc, long long cB, long long cH,
                    int K, int Hh, int flags)
{
    const int lane  = threadIdx.x & 31;
    const int wv    = threadIdx.x >> 5;
    const int bb    = blockIdx.z / Hh;
    const int hh    = blockIdx.z - bb * Hh;
    const int tileM = blockIdx.y * 128 + wv * 32;
    const int tileN = blockIdx.x * 64;

    A  += (long long)bb * aB + (long long)hh * aH;
    Bt += (long long)bb * bB + (long long)hh * bH;
    const long long cOff = (long long)bb * cB + (long long)hh * cH;

    // A fragment addressing (16-bit A 16x32, contiguous k order, same as B):
    // lane L: row m = base + (L&15); K chunk [kb..kb+15], kb = (L>=16)*16
    const int  kOffA = (lane >> 4) * 16;
    const bf16* arow0 = A + (long long)(tileM + (lane & 15)) * lda + kOffA;
    const bf16* arow1 = arow0 + (long long)16 * lda;

    // B fragment addressing (B 32x16, contiguous k order as A, via transposed operand):
    // lane L: col n = tileN + t*16 + (L&15); K chunk [kb2..kb2+15], kb2 = (L>=16)*16
    const int  nlane = lane & 15;
    const int  kOffB = (lane >> 4) * 16;
    const bf16* brow = Bt + (long long)(tileN + nlane) * ldb + kOffB;
    const long long bstep = (long long)16 * ldb;   // next 16-col tile

    v8f acc00 = {}, acc01 = {}, acc02 = {}, acc03 = {};
    v8f acc10 = {}, acc11 = {}, acc12 = {}, acc13 = {};

    for (int k0 = 0; k0 < K; k0 += 32) {
        __builtin_prefetch(arow0 + k0 + 64, 0, 1);
        __builtin_prefetch(arow1 + k0 + 64, 0, 1);
        __builtin_prefetch(brow  + k0 + 64, 0, 1);

        v8bf a0lo = *(const v8bf*)(arow0 + k0);
        v8bf a0hi = *(const v8bf*)(arow0 + k0 + 8);
        bf16x16 a0t{a0lo, a0hi};
        v16bf af0 = __builtin_bit_cast(v16bf, a0t);

        v8bf a1lo = *(const v8bf*)(arow1 + k0);
        v8bf a1hi = *(const v8bf*)(arow1 + k0 + 8);
        bf16x16 a1t{a1lo, a1hi};
        v16bf af1 = __builtin_bit_cast(v16bf, a1t);

        v16bf b0 = *(const v16bf*)(brow + k0);
        v16bf b1 = *(const v16bf*)(brow + bstep + k0);
        v16bf b2 = *(const v16bf*)(brow + 2 * bstep + k0);
        v16bf b3 = *(const v16bf*)(brow + 3 * bstep + k0);

        acc00 = __builtin_amdgcn_wmma_f32_16x16x32_bf16(false, af0, false, b0, (short)0, acc00, false, false);
        acc01 = __builtin_amdgcn_wmma_f32_16x16x32_bf16(false, af0, false, b1, (short)0, acc01, false, false);
        acc02 = __builtin_amdgcn_wmma_f32_16x16x32_bf16(false, af0, false, b2, (short)0, acc02, false, false);
        acc03 = __builtin_amdgcn_wmma_f32_16x16x32_bf16(false, af0, false, b3, (short)0, acc03, false, false);
        acc10 = __builtin_amdgcn_wmma_f32_16x16x32_bf16(false, af1, false, b0, (short)0, acc10, false, false);
        acc11 = __builtin_amdgcn_wmma_f32_16x16x32_bf16(false, af1, false, b1, (short)0, acc11, false, false);
        acc12 = __builtin_amdgcn_wmma_f32_16x16x32_bf16(false, af1, false, b2, (short)0, acc12, false, false);
        acc13 = __builtin_amdgcn_wmma_f32_16x16x32_bf16(false, af1, false, b3, (short)0, acc13, false, false);
    }

    // C layout: lane L holds col n = (L&15); VGPR r holds row r + (L>=16)*8
    v8f accs[2][4] = {{acc00, acc01, acc02, acc03}, {acc10, acc11, acc12, acc13}};
    #pragma unroll
    for (int s = 0; s < 2; ++s) {
        const int mBase = tileM + s * 16 + ((lane >> 4) << 3);
        #pragma unroll
        for (int t = 0; t < 4; ++t) {
            const int nn = tileN + t * 16 + nlane;
            #pragma unroll
            for (int r = 0; r < 8; ++r) {
                float v = accs[s][t][r];
                if (flags & GF_RELU) v = v > 0.0f ? v : 0.0f;
                const int mm = mBase + r;
                const long long idx = (flags & GF_TRANS)
                    ? ((long long)nn * ldc + mm)
                    : ((long long)mm * ldc + nn);
                if (flags & GF_OUTBF) ((bf16*)Cv)[cOff + idx] = (bf16)v;
                else                  ((float*)Cv)[cOff + idx] = v;
            }
        }
    }
}

// ---------------------------------------------------------------------------
// f32 -> bf16 elementwise cast
// ---------------------------------------------------------------------------
__global__ __launch_bounds__(256)
void cast_f32_bf16(const float* __restrict__ in, bf16* __restrict__ out, long long n)
{
    long long i = (long long)blockIdx.x * blockDim.x + threadIdx.x;
    const long long stride = (long long)gridDim.x * blockDim.x;
    for (; i < n; i += stride) out[i] = (bf16)in[i];
}

// ---------------------------------------------------------------------------
// W (KxN f32, row-major) -> Wt (NxK bf16, row-major)   (weight transpose+cast)
// ---------------------------------------------------------------------------
__global__ __launch_bounds__(256)
void transpose_cast(const float* __restrict__ W, bf16* __restrict__ Wt, int K, int N)
{
    const long long n = (long long)K * N;
    long long i = (long long)blockIdx.x * blockDim.x + threadIdx.x;
    const long long stride = (long long)gridDim.x * blockDim.x;
    for (; i < n; i += stride) {
        const int k = (int)(i / N);
        const int c = (int)(i - (long long)k * N);
        Wt[(long long)c * K + k] = (bf16)W[i];
    }
}

// ---------------------------------------------------------------------------
// LayerNorm over last dim (D=512), optional residual; dual f32/bf16 output.
// One block (256 threads) per row, 2 elements per thread.
// ---------------------------------------------------------------------------
__global__ __launch_bounds__(256)
void ln_rows(const float* __restrict__ x, const float* __restrict__ res,
             float* __restrict__ of, bf16* __restrict__ ob, int Dc)
{
    __shared__ float sred[256];
    const long long row = blockIdx.x;
    const int tid = threadIdx.x;
    const float* xr = x + row * Dc;
    float v0 = xr[tid];
    float v1 = xr[tid + 256];
    if (res) { v0 += res[row * Dc + tid]; v1 += res[row * Dc + tid + 256]; }

    sred[tid] = v0 + v1; __syncthreads();
    for (int off = 128; off > 0; off >>= 1) {
        if (tid < off) sred[tid] += sred[tid + off];
        __syncthreads();
    }
    const float mean = sred[0] * (1.0f / Dc);
    __syncthreads();

    const float d0 = v0 - mean, d1 = v1 - mean;
    sred[tid] = d0 * d0 + d1 * d1; __syncthreads();
    for (int off = 128; off > 0; off >>= 1) {
        if (tid < off) sred[tid] += sred[tid + off];
        __syncthreads();
    }
    const float rstd = rsqrtf(sred[0] * (1.0f / Dc) + EPS_);

    const float y0 = d0 * rstd, y1 = d1 * rstd;
    if (of) { of[row * Dc + tid] = y0; of[row * Dc + tid + 256] = y1; }
    if (ob) { ob[row * Dc + tid] = (bf16)y0; ob[row * Dc + tid + 256] = (bf16)y1; }
}

// ---------------------------------------------------------------------------
// Scaled softmax over rows of length 1024; dual f32(optional)/bf16 output.
// One block (256 threads) per row, 4 elements per thread.
// ---------------------------------------------------------------------------
__global__ __launch_bounds__(256)
void softmax_rows(const float* __restrict__ in, float scale,
                  float* __restrict__ of, bf16* __restrict__ ob)
{
    __shared__ float sred[256];
    const long long row = blockIdx.x;
    const int tid = threadIdx.x;
    const float* xr = in + row * 1024;

    float v[4];
    float mx = -3.4e38f;
    #pragma unroll
    for (int j = 0; j < 4; ++j) {
        v[j] = xr[tid + j * 256] * scale;
        mx = fmaxf(mx, v[j]);
    }
    sred[tid] = mx; __syncthreads();
    for (int off = 128; off > 0; off >>= 1) {
        if (tid < off) sred[tid] = fmaxf(sred[tid], sred[tid + off]);
        __syncthreads();
    }
    mx = sred[0]; __syncthreads();

    float sum = 0.0f;
    #pragma unroll
    for (int j = 0; j < 4; ++j) { v[j] = __expf(v[j] - mx); sum += v[j]; }
    sred[tid] = sum; __syncthreads();
    for (int off = 128; off > 0; off >>= 1) {
        if (tid < off) sred[tid] += sred[tid + off];
        __syncthreads();
    }
    const float inv = 1.0f / sred[0];

    #pragma unroll
    for (int j = 0; j < 4; ++j) {
        const float y = v[j] * inv;
        const long long o = row * 1024 + tid + j * 256;
        if (of) of[o] = y;
        ob[o] = (bf16)y;
    }
}

// ---------------------------------------------------------------------------
// Host orchestration
// ---------------------------------------------------------------------------
extern "C" void kernel_launch(void* const* d_in, const int* in_sizes, int n_in,
                              void* d_out, int out_size, void* d_ws, size_t ws_size,
                              hipStream_t stream)
{
    (void)in_sizes; (void)n_in; (void)out_size; (void)ws_size;

    char* ws = (char*)d_ws;
    size_t off = 0;
    auto alloc = [&](size_t bytes) -> void* {
        off = (off + 255) & ~(size_t)255;
        void* p = ws + off;
        off += bytes;
        return p;
    };

    auto gemm = [&](const void* A, int lda, long long aB, long long aH,
                    const void* Bt, int ldb, long long bB, long long bH,
                    void* C, int ldc, long long cB, long long cH,
                    int M, int N, int K, int batches, int Hh, int flags) {
        dim3 g((unsigned)(N / 64), (unsigned)(M / 128), (unsigned)batches);
        gemm_bf16_wmma<<<g, 128, 0, stream>>>((const bf16*)A, lda, aB, aH,
                                              (const bf16*)Bt, ldb, bB, bH,
                                              C, ldc, cB, cH, K, Hh, flags);
    };

    // ---- weight transpose+cast (inputs 2..19) ----
    bf16* wT[20] = {};
    for (int i = 2; i < 20; ++i) {
        int K, N;
        if (i < 12)            { K = D_;   N = D_;   }
        else if ((i & 1) == 0) { K = D_;   N = DFF_; }  // F*1
        else                   { K = DFF_; N = D_;   }  // F*2
        wT[i] = (bf16*)alloc((size_t)K * N * sizeof(bf16));
        transpose_cast<<<2048, 256, 0, stream>>>((const float*)d_in[i], wT[i], K, N);
    }
    bf16 *WqT = wT[2],  *WkT = wT[3],  *WvT = wT[4],  *WoT = wT[5];
    bf16 *AqT = wT[6],  *AkT = wT[7],  *AvaT = wT[8], *AvbT = wT[9];
    bf16 *AfaT = wT[10], *AfbT = wT[11];
    bf16 *Fa1T = wT[12], *Fa2T = wT[13], *Fb1T = wT[14], *Fb2T = wT[15];
    bf16 *Faa1T = wT[16], *Faa2T = wT[17], *Fbb1T = wT[18], *Fbb2T = wT[19];

    // ---- activations cast ----
    const long long ND = (long long)BS_ * D_;
    bf16* Xa = (bf16*)alloc(ND * 2);
    bf16* Xb = (bf16*)alloc(ND * 2);
    cast_f32_bf16<<<4096, 256, 0, stream>>>((const float*)d_in[0], Xa, ND);
    cast_f32_bf16<<<4096, 256, 0, stream>>>((const float*)d_in[1], Xb, ND);

    // ---- work buffers ----
    bf16*  Qb     = (bf16*)alloc(ND * 2);
    bf16*  Kb     = (bf16*)alloc(ND * 2);
    bf16*  Vt     = (bf16*)alloc(ND * 2);              // transposed V (512 x 8192)
    bf16*  Vt2    = (bf16*)alloc(ND * 2);
    float* scores = (float*)alloc((size_t)B_ * H_ * S_ * S_ * 4);   // 256 MB
    bf16*  attnb  = (bf16*) alloc((size_t)B_ * H_ * S_ * S_ * 2);   // 128 MB
    bf16*  ctx1   = (bf16*)alloc(ND * 2);
    bf16*  ctx2   = (bf16*)alloc(ND * 2);
    float* proj   = (float*)alloc(ND * 4);
    bf16*  hid    = (bf16*)alloc((size_t)BS_ * DFF_ * 2);
    float* Y1f    = (float*)alloc(ND * 4);
    bf16*  Y1b    = (bf16*)alloc(ND * 2);
    float* AQf    = (float*)alloc(ND * 4);
    bf16*  AQb    = (bf16*)alloc(ND * 2);
    float* KBf    = (float*)alloc(ND * 4);
    bf16*  KBb    = (bf16*)alloc(ND * 2);
    float* OAf    = (float*)alloc(ND * 4);
    bf16*  OAb    = (bf16*)alloc(ND * 2);
    float* OBf    = (float*)alloc(ND * 4);
    bf16*  OBb    = (bf16*)alloc(ND * 2);

    const long long zS  = (long long)S_ * D_;     // A/B per-batch stride for (b,h) blocks
    const long long zSS = (long long)S_ * S_;
    const long long zHSS = (long long)H_ * zSS;

    // ---- self-attention + FFN for one stream (shared Wq/Wk/Wv/Wo) ----
    auto self_stream = [&](const bf16* Xbf, const float* Xres,
                           bf16* F1T, bf16* F2T, float* Yf, bf16* Yb) {
        gemm(Xbf, D_, 0, 0, WqT, D_, 0, 0, Qb, D_, 0, 0, BS_, D_, D_, 1, 1, GF_OUTBF);
        gemm(Xbf, D_, 0, 0, WkT, D_, 0, 0, Kb, D_, 0, 0, BS_, D_, D_, 1, 1, GF_OUTBF);
        gemm(Xbf, D_, 0, 0, WvT, D_, 0, 0, Vt, BS_, 0, 0, BS_, D_, D_, 1, 1, GF_OUTBF | GF_TRANS);
        // scores[b,h] = Q_bh (SxDK) * K_bh^T
        gemm(Qb, D_, zS, DK_, Kb, D_, zS, DK_,
             scores, S_, zHSS, zSS, S_, S_, DK_, B_ * H_, H_, 0);
        softmax_rows<<<B_ * H_ * S_, 256, 0, stream>>>(scores, SCALE_, nullptr, attnb);
        // ctx[b,h] = attn_bh (SxS) * V_bh ; Vt rows = channel (h*64+d), ld = BS
        gemm(attnb, S_, zHSS, zSS,
             Vt, BS_, (long long)S_, (long long)DK_ * BS_,
             ctx1, D_, zS, DK_, S_, DK_, S_, B_ * H_, H_, GF_OUTBF);
        gemm(ctx1, D_, 0, 0, WoT, D_, 0, 0, proj, D_, 0, 0, BS_, D_, D_, 1, 1, 0);
        ln_rows<<<BS_, 256, 0, stream>>>(proj, Xres, Y1f, Y1b, D_);
        // FFN
        gemm(Y1b, D_, 0, 0, F1T, D_, 0, 0, hid, DFF_, 0, 0, BS_, DFF_, D_, 1, 1, GF_OUTBF | GF_RELU);
        gemm(hid, DFF_, 0, 0, F2T, DFF_, 0, 0, proj, D_, 0, 0, BS_, D_, DFF_, 1, 1, 0);
        ln_rows<<<BS_, 256, 0, stream>>>(proj, Y1f, Yf, Yb, D_);
    };

    self_stream(Xa, (const float*)d_in[0], Fa1T, Fa2T, AQf, AQb);   // action_query
    self_stream(Xb, (const float*)d_in[1], Fb1T, Fb2T, KBf, KBb);   // key_value_bg

    // ---- ABI cross-attention ----
    gemm(AQb, D_, 0, 0, AqT,  D_, 0, 0, Qb,  D_,  0, 0, BS_, D_, D_, 1, 1, GF_OUTBF);
    gemm(KBb, D_, 0, 0, AkT,  D_, 0, 0, Kb,  D_,  0, 0, BS_, D_, D_, 1, 1, GF_OUTBF);
    gemm(AQb, D_, 0, 0, AvaT, D_, 0, 0, Vt,  BS_, 0, 0, BS_, D_, D_, 1, 1, GF_OUTBF | GF_TRANS);
    gemm(KBb, D_, 0, 0, AvbT, D_, 0, 0, Vt2, BS_, 0, 0, BS_, D_, D_, 1, 1, GF_OUTBF | GF_TRANS);

    gemm(Qb, D_, zS, DK_, Kb, D_, zS, DK_,
         scores, S_, zHSS, zSS, S_, S_, DK_, B_ * H_, H_, 0);

    float* attn_out = (float*)d_out + 2LL * BS_ * D_;
    softmax_rows<<<B_ * H_ * S_, 256, 0, stream>>>(scores, SCALE_, attn_out, attnb);

    gemm(attnb, S_, zHSS, zSS, Vt,  BS_, (long long)S_, (long long)DK_ * BS_,
         ctx1, D_, zS, DK_, S_, DK_, S_, B_ * H_, H_, GF_OUTBF);
    gemm(attnb, S_, zHSS, zSS, Vt2, BS_, (long long)S_, (long long)DK_ * BS_,
         ctx2, D_, zS, DK_, S_, DK_, S_, B_ * H_, H_, GF_OUTBF);

    gemm(ctx1, D_, 0, 0, AfaT, D_, 0, 0, proj, D_, 0, 0, BS_, D_, D_, 1, 1, 0);
    ln_rows<<<BS_, 256, 0, stream>>>(proj, AQf, OAf, OAb, D_);
    gemm(ctx2, D_, 0, 0, AfbT, D_, 0, 0, proj, D_, 0, 0, BS_, D_, D_, 1, 1, 0);
    ln_rows<<<BS_, 256, 0, stream>>>(proj, KBf, OBf, OBb, D_);

    // ---- final FFNs -> d_out ----
    gemm(OAb, D_, 0, 0, Faa1T, D_, 0, 0, hid, DFF_, 0, 0, BS_, DFF_, D_, 1, 1, GF_OUTBF | GF_RELU);
    gemm(hid, DFF_, 0, 0, Faa2T, DFF_, 0, 0, proj, D_, 0, 0, BS_, D_, DFF_, 1, 1, 0);
    ln_rows<<<BS_, 256, 0, stream>>>(proj, OAf, (float*)d_out, nullptr, D_);

    gemm(OBb, D_, 0, 0, Fbb1T, D_, 0, 0, hid, DFF_, 0, 0, BS_, DFF_, D_, 1, 1, GF_OUTBF | GF_RELU);
    gemm(hid, DFF_, 0, 0, Fbb2T, DFF_, 0, 0, proj, D_, 0, 0, BS_, D_, DFF_, 1, 1, 0);
    ln_rows<<<BS_, 256, 0, stream>>>(proj, OBf, (float*)d_out + (long long)BS_ * D_, nullptr, D_);
}
